// GRViT_137438954052
// MI455X (gfx1250) — matrix operand order repaired and checked by reading, not compiled, not run on hardware
//
#include <hip/hip_runtime.h>
#include <math.h>

typedef _Float16 h16 __attribute__((ext_vector_type(16)));
typedef float f8 __attribute__((ext_vector_type(8)));

#define WMMA(a, b, c) __builtin_amdgcn_wmma_f32_16x16x32_f16(false, (a), false, (b), (short)0, (c), false, false)

static constexpr int B_ = 8, N_ = 4096, C_ = 256, H_ = 8, D_ = 32, HID_ = 1024;
static constexpr int M_ = B_ * N_; // 32768 rows

// ---------------- fragment loaders (f16, row-major TN GEMM) ----------------
// A: [M,K] row-major.  Lane r=lane&15 owns row m0+r; h=lane>>4 selects K-half.
__device__ __forceinline__ h16 load_a_frag(const _Float16* A, int lda, int m0, int k, int r, int h) {
    union { h16 v; uint4 q[2]; } u;
    const _Float16* p = A + (size_t)(m0 + r) * lda + k;
    u.q[0] = *reinterpret_cast<const uint4*>(p + h * 8);        // K = h*8 .. +7   -> elems 0..7
    u.q[1] = *reinterpret_cast<const uint4*>(p + 16 + h * 8);   // K = 16+h*8 ..   -> elems 8..15
    return u.v;
}
// B = W^T with W: [N,K] row-major.  Lane r owns column n0+r; K = h*8..+7 and 16+h*8..+7.
__device__ __forceinline__ h16 load_b_frag(const _Float16* W, int ldw, int n0, int k, int r, int h) {
    union { h16 v; uint4 q[2]; } u;
    const _Float16* p = W + (size_t)(n0 + r) * ldw + k;
    u.q[0] = *reinterpret_cast<const uint4*>(p + h * 8);
    u.q[1] = *reinterpret_cast<const uint4*>(p + 16 + h * 8);
    return u.v;
}
__device__ __forceinline__ f8 f8zero() {
    f8 z;
#pragma unroll
    for (int i = 0; i < 8; ++i) z[i] = 0.0f;
    return z;
}
// 32x64 output strip per wave: 2 M-tiles x 4 N-tiles; each B fragment feeds 2 WMMAs.
__device__ __forceinline__ void gemm_strip_32x64(const _Float16* A, int lda,
                                                 const _Float16* W, int ldw,
                                                 int m0, int n0, int K, f8 acc[2][4]) {
    const int lane = threadIdx.x & 31, r = lane & 15, h = lane >> 4;
    for (int k = 0; k < K; k += 32) {
        h16 a0 = load_a_frag(A, lda, m0, k, r, h);
        h16 a1 = load_a_frag(A, lda, m0 + 16, k, r, h);
#pragma unroll
        for (int nt = 0; nt < 4; ++nt) {
            h16 b = load_b_frag(W, ldw, n0 + nt * 16, k, r, h);
            acc[0][nt] = WMMA(a0, b, acc[0][nt]);
            acc[1][nt] = WMMA(a1, b, acc[1][nt]);
        }
    }
}

__device__ __forceinline__ float wred(float v) {
    v += __shfl_xor(v, 16, 32);
    v += __shfl_xor(v, 8, 32);
    v += __shfl_xor(v, 4, 32);
    v += __shfl_xor(v, 2, 32);
    v += __shfl_xor(v, 1, 32);
    return v;
}

// ---------------- weight f32 -> f16 ----------------
__global__ void f2h_kernel(const float* __restrict__ src, _Float16* __restrict__ dst, int n) {
    int i = blockIdx.x * blockDim.x + threadIdx.x;
    if (i < n) dst[i] = (_Float16)src[i];
}

// ---------------- prep: input_pos, LN1, LN2, gate-input staging ----------------
__global__ void prep_kernel(const float* __restrict__ input_, const float* __restrict__ prev,
                            const float* __restrict__ pos,
                            const float* __restrict__ n1w, const float* __restrict__ n1b,
                            const float* __restrict__ n2w, const float* __restrict__ n2b,
                            float* __restrict__ IP32, _Float16* __restrict__ X16,
                            _Float16* __restrict__ S16, _Float16* __restrict__ G16) {
    const int m = (blockIdx.x * blockDim.x + threadIdx.x) >> 5;   // one wave per row
    const int lane = threadIdx.x & 31;
    const int n = m & (N_ - 1);
    const int cb = lane * 8;
    const size_t rb = (size_t)m * C_ + cb;
    float4 i0 = *(const float4*)(input_ + rb), i1 = *(const float4*)(input_ + rb + 4);
    float4 p0 = *(const float4*)(pos + (size_t)n * C_ + cb), p1 = *(const float4*)(pos + (size_t)n * C_ + cb + 4);
    float4 s0 = *(const float4*)(prev + rb), s1 = *(const float4*)(prev + rb + 4);
    float ri[8] = {i0.x, i0.y, i0.z, i0.w, i1.x, i1.y, i1.z, i1.w};
    float rp[8] = {p0.x, p0.y, p0.z, p0.w, p1.x, p1.y, p1.z, p1.w};
    float rs_[8] = {s0.x, s0.y, s0.z, s0.w, s1.x, s1.y, s1.z, s1.w};
    float xi[8];
#pragma unroll
    for (int i = 0; i < 8; ++i) xi[i] = ri[i] + rp[i];
    *(float4*)(IP32 + rb) = make_float4(xi[0], xi[1], xi[2], xi[3]);
    *(float4*)(IP32 + rb + 4) = make_float4(xi[4], xi[5], xi[6], xi[7]);
    // LN1(input_pos)
    float s = 0.f, sq = 0.f;
#pragma unroll
    for (int i = 0; i < 8; ++i) { s += xi[i]; sq += xi[i] * xi[i]; }
    s = wred(s); sq = wred(sq);
    float mu = s * (1.f / C_), var = sq * (1.f / C_) - mu * mu, inv = rsqrtf(var + 1e-5f);
#pragma unroll
    for (int i = 0; i < 8; ++i)
        X16[rb + i] = (_Float16)((xi[i] - mu) * inv * n1w[cb + i] + n1b[cb + i]);
    // LN2(prev_state)
    s = 0.f; sq = 0.f;
#pragma unroll
    for (int i = 0; i < 8; ++i) { s += rs_[i]; sq += rs_[i] * rs_[i]; }
    s = wred(s); sq = wred(sq);
    mu = s * (1.f / C_); var = sq * (1.f / C_) - mu * mu; inv = rsqrtf(var + 1e-5f);
#pragma unroll
    for (int i = 0; i < 8; ++i)
        S16[rb + i] = (_Float16)((rs_[i] - mu) * inv * n2w[cb + i] + n2b[cb + i]);
    // gate input: stacked = [input_, prev_state]  -> [M, 512] f16
    const size_t gb = (size_t)m * (2 * C_) + cb;
#pragma unroll
    for (int i = 0; i < 8; ++i) { G16[gb + i] = (_Float16)ri[i]; G16[gb + C_ + i] = (_Float16)rs_[i]; }
}

// ---------------- qkv: out = x_ln@Wi^T + s_ln@Ws^T, phi epilogue + transposed scatter ----------------
__global__ void qkv_kernel(const _Float16* __restrict__ X16, const _Float16* __restrict__ S16,
                           const _Float16* __restrict__ Wi, const _Float16* __restrict__ Ws,
                           _Float16* __restrict__ PQ,   // [B,H,N,D]
                           _Float16* __restrict__ PKt,  // [B,H,D,N]
                           _Float16* __restrict__ Vt) { // [B,H,D,N]
    const int task = (blockIdx.x * blockDim.x + threadIdx.x) >> 5;  // 1024*12 waves
    const int ms = task / 12, ns = task % 12;
    const int m0 = ms * 32, n0 = ns * 64;
    f8 acc[2][4] = {{f8zero(), f8zero(), f8zero(), f8zero()},
                    {f8zero(), f8zero(), f8zero(), f8zero()}};
    gemm_strip_32x64(X16, C_, Wi, C_, m0, n0, C_, acc);
    gemm_strip_32x64(S16, C_, Ws, C_, m0, n0, C_, acc);
    const int lane = threadIdx.x & 31, r = lane & 15, h = lane >> 4;
#pragma unroll
    for (int mt = 0; mt < 2; ++mt)
#pragma unroll
        for (int nt = 0; nt < 4; ++nt)
#pragma unroll
            for (int j = 0; j < 8; ++j) {
                const int row = m0 + mt * 16 + j + 8 * h;  // b*N + n
                const int col = n0 + nt * 16 + r;          // s*256 + hh*32 + d
                const int b = row >> 12, n = row & (N_ - 1);
                const int sct = col >> 8, rem = col & 255, hh = rem >> 5, d = rem & 31;
                const float v = acc[mt][nt][j];
                if (sct == 0) {
                    const float pv = v > 0.f ? v + 1.f : expf(v);          // elu(v)+1
                    PQ[((size_t)(b * H_ + hh) * N_ + n) * D_ + d] = (_Float16)pv;
                } else if (sct == 1) {
                    const float pv = v > 0.f ? v + 1.f : expf(v);
                    PKt[((size_t)(b * H_ + hh) * D_ + d) * N_ + n] = (_Float16)pv;
                } else {
                    Vt[((size_t)(b * H_ + hh) * D_ + d) * N_ + n] = (_Float16)v;
                }
            }
}

// ---------------- kv[d,e] = sum_n phi_k[n,d] v[n,e], per (b,h); store kv^T ----------------
__global__ void kv_kernel(const _Float16* __restrict__ PKt, const _Float16* __restrict__ Vt,
                          _Float16* __restrict__ KVt) {   // [BH, e, d] = [BH,32,32]
    const int bh = blockIdx.x;                 // 64 blocks, 4 waves each
    const int wave = threadIdx.x >> 5;
    const int m0 = (wave >> 1) * 16, n0 = (wave & 1) * 16;
    const _Float16* A = PKt + (size_t)bh * D_ * N_;   // A[d, n], lda = 4096
    const _Float16* W = Vt + (size_t)bh * D_ * N_;    // W[e, n], ldw = 4096
    const int lane = threadIdx.x & 31, r = lane & 15, h = lane >> 4;
    f8 acc = f8zero();
    for (int k = 0; k < N_; k += 32) {
        h16 a = load_a_frag(A, N_, m0, k, r, h);
        h16 b = load_b_frag(W, N_, n0, k, r, h);
        acc = WMMA(a, b, acc);
    }
#pragma unroll
    for (int j = 0; j < 8; ++j) {
        const int d = m0 + j + 8 * h, e = n0 + r;
        KVt[(size_t)bh * (D_ * D_) + e * D_ + d] = (_Float16)acc[j];
    }
}

// ---------------- attn[n,e] = sum_d phi_q[n,d]*kv[d,e]; write [M, C] for proj ----------------
__global__ void attn_kernel(const _Float16* __restrict__ PQ, const _Float16* __restrict__ KVt,
                            _Float16* __restrict__ AT16) {
    const int task = (blockIdx.x * blockDim.x + threadIdx.x) >> 5;  // 64*128 waves
    const int bh = task >> 7, ms = task & 127;
    const int m0 = ms * 32;
    const _Float16* A = PQ + (size_t)bh * N_ * D_;    // [n, d], lda = 32
    const _Float16* W = KVt + (size_t)bh * (D_ * D_); // W[e, d], ldw = 32
    const int lane = threadIdx.x & 31, r = lane & 15, h = lane >> 4;
    f8 acc[2][2] = {{f8zero(), f8zero()}, {f8zero(), f8zero()}};
    h16 a0 = load_a_frag(A, D_, m0, 0, r, h);
    h16 a1 = load_a_frag(A, D_, m0 + 16, 0, r, h);
#pragma unroll
    for (int nt = 0; nt < 2; ++nt) {
        h16 b = load_b_frag(W, D_, nt * 16, 0, r, h);
        acc[0][nt] = WMMA(a0, b, acc[0][nt]);
        acc[1][nt] = WMMA(a1, b, acc[1][nt]);
    }
    const int b_ = bh >> 3, hh = bh & 7;
#pragma unroll
    for (int mt = 0; mt < 2; ++mt)
#pragma unroll
        for (int nt = 0; nt < 2; ++nt)
#pragma unroll
            for (int j = 0; j < 8; ++j) {
                const int n = m0 + mt * 16 + j + 8 * h, e = nt * 16 + r;
                AT16[((size_t)(b_ * N_ + n)) * C_ + hh * D_ + e] = (_Float16)acc[mt][nt][j];
            }
}

// ---------------- proj: out0 = input_pos + (attn@Pw^T + pb); IP32 <- attn_proj ----------------
__global__ void proj_kernel(const _Float16* __restrict__ AT16, const _Float16* __restrict__ Pw,
                            const float* __restrict__ pb, float* __restrict__ IP32,
                            float* __restrict__ out0) {
    const int task = (blockIdx.x * blockDim.x + threadIdx.x) >> 5;  // 1024*4
    const int m0 = (task >> 2) * 32, n0 = (task & 3) * 64;
    f8 acc[2][4] = {{f8zero(), f8zero(), f8zero(), f8zero()},
                    {f8zero(), f8zero(), f8zero(), f8zero()}};
    gemm_strip_32x64(AT16, C_, Pw, C_, m0, n0, C_, acc);
    const int lane = threadIdx.x & 31, r = lane & 15, h = lane >> 4;
#pragma unroll
    for (int mt = 0; mt < 2; ++mt)
#pragma unroll
        for (int nt = 0; nt < 4; ++nt)
#pragma unroll
            for (int j = 0; j < 8; ++j) {
                const int row = m0 + mt * 16 + j + 8 * h, col = n0 + nt * 16 + r;
                const size_t idx = (size_t)row * C_ + col;
                const float v = acc[mt][nt][j] + pb[col];
                out0[idx] = IP32[idx] + v;
                IP32[idx] = v;   // attn (post-proj) for gate blend
            }
}

// ---------------- gate: new_state = prev*(1-u) + attn*u ----------------
__global__ void gate_kernel(const _Float16* __restrict__ G16, const _Float16* __restrict__ Gw,
                            const float* __restrict__ gb, const float* __restrict__ prev,
                            const float* __restrict__ AP32, float* __restrict__ out1) {
    const int task = (blockIdx.x * blockDim.x + threadIdx.x) >> 5;  // 1024*4
    const int m0 = (task >> 2) * 32, n0 = (task & 3) * 64;
    f8 acc[2][4] = {{f8zero(), f8zero(), f8zero(), f8zero()},
                    {f8zero(), f8zero(), f8zero(), f8zero()}};
    gemm_strip_32x64(G16, 2 * C_, Gw, 2 * C_, m0, n0, 2 * C_, acc);
    const int lane = threadIdx.x & 31, r = lane & 15, h = lane >> 4;
#pragma unroll
    for (int mt = 0; mt < 2; ++mt)
#pragma unroll
        for (int nt = 0; nt < 4; ++nt)
#pragma unroll
            for (int j = 0; j < 8; ++j) {
                const int row = m0 + mt * 16 + j + 8 * h, col = n0 + nt * 16 + r;
                const size_t idx = (size_t)row * C_ + col;
                const float u = 1.f / (1.f + expf(-(acc[mt][nt][j] + gb[col])));
                out1[idx] = prev[idx] * (1.f - u) + AP32[idx] * u;
            }
}

// ---------------- LN3 over out0 -> f16 ----------------
__global__ void ln3_kernel(const float* __restrict__ out0, const float* __restrict__ w,
                           const float* __restrict__ b, _Float16* __restrict__ H16) {
    const int m = (blockIdx.x * blockDim.x + threadIdx.x) >> 5;
    const int lane = threadIdx.x & 31, cb = lane * 8;
    const size_t rb = (size_t)m * C_ + cb;
    float4 a0 = *(const float4*)(out0 + rb), a1 = *(const float4*)(out0 + rb + 4);
    float x[8] = {a0.x, a0.y, a0.z, a0.w, a1.x, a1.y, a1.z, a1.w};
    float s = 0.f, sq = 0.f;
#pragma unroll
    for (int i = 0; i < 8; ++i) { s += x[i]; sq += x[i] * x[i]; }
    s = wred(s); sq = wred(sq);
    const float mu = s * (1.f / C_), var = sq * (1.f / C_) - mu * mu, inv = rsqrtf(var + 1e-5f);
#pragma unroll
    for (int i = 0; i < 8; ++i)
        H16[rb + i] = (_Float16)((x[i] - mu) * inv * w[cb + i] + b[cb + i]);
}

// ---------------- fc1: GELU(h@W1^T + b1) -> f16 [M,HID] ----------------
__global__ void fc1_kernel(const _Float16* __restrict__ H16, const _Float16* __restrict__ W1,
                           const float* __restrict__ b1, _Float16* __restrict__ H1) {
    const int task = (blockIdx.x * blockDim.x + threadIdx.x) >> 5;  // 1024*16
    const int m0 = (task >> 4) * 32, n0 = (task & 15) * 64;
    f8 acc[2][4] = {{f8zero(), f8zero(), f8zero(), f8zero()},
                    {f8zero(), f8zero(), f8zero(), f8zero()}};
    gemm_strip_32x64(H16, C_, W1, C_, m0, n0, C_, acc);
    const int lane = threadIdx.x & 31, r = lane & 15, h = lane >> 4;
#pragma unroll
    for (int mt = 0; mt < 2; ++mt)
#pragma unroll
        for (int nt = 0; nt < 4; ++nt)
#pragma unroll
            for (int j = 0; j < 8; ++j) {
                const int row = m0 + mt * 16 + j + 8 * h, col = n0 + nt * 16 + r;
                const float x = acc[mt][nt][j] + b1[col];
                const float g = 0.5f * x * (1.f + erff(x * 0.70710678118654752f));  // exact gelu
                H1[(size_t)row * HID_ + col] = (_Float16)g;
            }
}

// ---------------- fc2: out0 += h1@W2^T + b2 ----------------
__global__ void fc2_kernel(const _Float16* __restrict__ H1, const _Float16* __restrict__ W2,
                           const float* __restrict__ b2, float* __restrict__ out0) {
    const int task = (blockIdx.x * blockDim.x + threadIdx.x) >> 5;  // 1024*4
    const int m0 = (task >> 2) * 32, n0 = (task & 3) * 64;
    f8 acc[2][4] = {{f8zero(), f8zero(), f8zero(), f8zero()},
                    {f8zero(), f8zero(), f8zero(), f8zero()}};
    gemm_strip_32x64(H1, HID_, W2, HID_, m0, n0, HID_, acc);
    const int lane = threadIdx.x & 31, r = lane & 15, h = lane >> 4;
#pragma unroll
    for (int mt = 0; mt < 2; ++mt)
#pragma unroll
        for (int nt = 0; nt < 4; ++nt)
#pragma unroll
            for (int j = 0; j < 8; ++j) {
                const int row = m0 + mt * 16 + j + 8 * h, col = n0 + nt * 16 + r;
                const size_t idx = (size_t)row * C_ + col;
                out0[idx] = out0[idx] + acc[mt][nt][j] + b2[col];
            }
}

// ---------------- workspace layout (bytes) ----------------
static constexpr size_t OFF_IP  = 0;                          // f32 [M,C]  input_pos -> attn_proj
static constexpr size_t OFF_X16 = OFF_IP  + (size_t)M_ * C_ * 4;      // f16 [M,C]  x_ln -> attn (reuse)
static constexpr size_t OFF_S16 = OFF_X16 + (size_t)M_ * C_ * 2;      // f16 [M,C]  s_ln -> ln3 h (reuse)
static constexpr size_t OFF_G16 = OFF_S16 + (size_t)M_ * C_ * 2;      // f16 [M,2C] stacked gate input
static constexpr size_t OFF_PQ  = OFF_G16 + (size_t)M_ * 2 * C_ * 2;  // f16 [B,H,N,D]
static constexpr size_t OFF_PKT = OFF_PQ  + (size_t)M_ * C_ * 2;      // f16 [B,H,D,N]
static constexpr size_t OFF_VT  = OFF_PKT + (size_t)M_ * C_ * 2;      // f16 [B,H,D,N]
static constexpr size_t OFF_KVT = OFF_VT  + (size_t)M_ * C_ * 2;      // f16 [BH,32,32]
static constexpr size_t OFF_H1  = OFF_KVT + (size_t)B_ * H_ * D_ * D_ * 2; // f16 [M,HID]
static constexpr size_t OFF_W   = OFF_H1  + (size_t)M_ * HID_ * 2;    // f16 weights
static constexpr size_t WSZ_QKV = (size_t)3 * C_ * C_;       // 196608 elements
static constexpr size_t OFF_WI  = OFF_W;
static constexpr size_t OFF_WS2 = OFF_WI  + WSZ_QKV * 2;
static constexpr size_t OFF_PW  = OFF_WS2 + WSZ_QKV * 2;
static constexpr size_t OFF_GW  = OFF_PW  + (size_t)C_ * C_ * 2;
static constexpr size_t OFF_W1  = OFF_GW  + (size_t)C_ * 2 * C_ * 2;
static constexpr size_t OFF_W2  = OFF_W1  + (size_t)HID_ * C_ * 2;

extern "C" void kernel_launch(void* const* d_in, const int* in_sizes, int n_in,
                              void* d_out, int out_size, void* d_ws, size_t ws_size,
                              hipStream_t stream) {
    const float* input_  = (const float*)d_in[0];
    const float* prev    = (const float*)d_in[1];
    const float* pos     = (const float*)d_in[2];
    const float* n1w = (const float*)d_in[3],  *n1b = (const float*)d_in[4];
    const float* n2w = (const float*)d_in[5],  *n2b = (const float*)d_in[6];
    const float* n3w = (const float*)d_in[7],  *n3b = (const float*)d_in[8];
    const float* qkv_iw = (const float*)d_in[9];
    const float* qkv_sw = (const float*)d_in[10];
    const float* proj_w = (const float*)d_in[11], *proj_b = (const float*)d_in[12];
    const float* gate_w = (const float*)d_in[13], *gate_b = (const float*)d_in[14];
    const float* fc1_w  = (const float*)d_in[15], *fc1_b  = (const float*)d_in[16];
    const float* fc2_w  = (const float*)d_in[17], *fc2_b  = (const float*)d_in[18];

    float* out0 = (float*)d_out;                      // output  [B,N,C]
    float* out1 = out0 + (size_t)M_ * C_;             // new_state [B,N,C]
    char* ws = (char*)d_ws;
    float*     IP32 = (float*)(ws + OFF_IP);
    _Float16*  X16  = (_Float16*)(ws + OFF_X16);
    _Float16*  S16  = (_Float16*)(ws + OFF_S16);
    _Float16*  G16  = (_Float16*)(ws + OFF_G16);
    _Float16*  PQ   = (_Float16*)(ws + OFF_PQ);
    _Float16*  PKT  = (_Float16*)(ws + OFF_PKT);
    _Float16*  VT   = (_Float16*)(ws + OFF_VT);
    _Float16*  KVT  = (_Float16*)(ws + OFF_KVT);
    _Float16*  H1   = (_Float16*)(ws + OFF_H1);
    _Float16*  Wi16 = (_Float16*)(ws + OFF_WI);
    _Float16*  Ws16 = (_Float16*)(ws + OFF_WS2);
    _Float16*  Pw16 = (_Float16*)(ws + OFF_PW);
    _Float16*  Gw16 = (_Float16*)(ws + OFF_GW);
    _Float16*  W116 = (_Float16*)(ws + OFF_W1);
    _Float16*  W216 = (_Float16*)(ws + OFF_W2);

    auto cvt = [&](const float* s, _Float16* d, int n) {
        f2h_kernel<<<(n + 255) / 256, 256, 0, stream>>>(s, d, n);
    };
    cvt(qkv_iw, Wi16, 3 * C_ * C_);
    cvt(qkv_sw, Ws16, 3 * C_ * C_);
    cvt(proj_w, Pw16, C_ * C_);
    cvt(gate_w, Gw16, C_ * 2 * C_);
    cvt(fc1_w,  W116, HID_ * C_);
    cvt(fc2_w,  W216, C_ * HID_);

    prep_kernel<<<M_ / 8, 256, 0, stream>>>(input_, prev, pos, n1w, n1b, n2w, n2b,
                                            IP32, X16, S16, G16);
    // qkv: 1024 m-strips (32 rows) * 12 n-strips (64 cols), 8 waves/block
    qkv_kernel<<<(M_ / 32) * 12 / 8, 256, 0, stream>>>(X16, S16, Wi16, Ws16, PQ, PKT, VT);
    kv_kernel<<<B_ * H_, 128, 0, stream>>>(PKT, VT, KVT);
    attn_kernel<<<(B_ * H_) * (N_ / 32) / 8, 256, 0, stream>>>(PQ, KVT, X16 /*AT16 reuses X16*/);
    proj_kernel<<<(M_ / 32) * 4 / 8, 256, 0, stream>>>(X16, Pw16, proj_b, IP32, out0);
    gate_kernel<<<(M_ / 32) * 4 / 8, 256, 0, stream>>>(G16, Gw16, gate_b, prev, IP32, out1);
    ln3_kernel<<<M_ / 8, 256, 0, stream>>>(out0, n3w, n3b, S16 /*H16 reuses S16*/);
    fc1_kernel<<<(M_ / 32) * 16 / 8, 256, 0, stream>>>(S16, W116, fc1_b, H1);
    fc2_kernel<<<(M_ / 32) * 4 / 8, 256, 0, stream>>>(H1, W216, fc2_b, out0);
}
